// ChannelAttention3D_37976100831786
// MI455X (gfx1250) — matrix operand order repaired and checked by reading, not compiled, not run on hardware
//
#include <hip/hip_runtime.h>
#include <hip/hip_bf16.h>

typedef _Float16 half_t;
typedef __attribute__((ext_vector_type(16))) _Float16 v16h;
typedef __attribute__((ext_vector_type(8)))  float    v8f;
typedef __attribute__((ext_vector_type(4)))  float    v4f;

// Problem dims
#define BATCH 2
#define CIN   32
#define DD    64
#define HH    128
#define WW    128

// Padded f16 x_cat buffer: [B][2][DP][HP][WP], halos pre-zeroed.
//  d index = DLEAD + d' (DLEAD=3), h index = HLEAD + h' (3), w index = WLEAD + w' (4).
// Conv B-fragment reads start at half-index w0 (j=0 -> input w = w0-4), so all
// v16h loads are 32-byte aligned. Toeplitz A has zeros at j=0 and j>=23.
#define DP 70      // 3 + 64 + 3
#define HP 134     // 3 + 128 + 3
#define WP 144     // 4 + 128 + 12  (max read half-index w0+31 = 143)
#define DLEAD 3
#define HLEAD 3
#define WLEAD 4

#define XCP_HALFS ((size_t)BATCH * 2 * DP * HP * WP)   // 5,402,880 halves
#define XCP_BYTES (XCP_HALFS * 2)                      // 10,805,760 B
#define NFRAG 98                                       // 2 ch * 7 kd * 7 kh
#define FRAG_BYTES ((size_t)NFRAG * 32 * 32)           // 100,352 B

// ---------------------------------------------------------------------------
// Stage 1: channel mean+max reduction, fp32 -> f16, into padded buffer.
// Read-once 268 MB stream: nontemporal loads keep L2 free for x_cat.
// ---------------------------------------------------------------------------
__global__ void __launch_bounds__(256)
sa_reduce_kernel(const float* __restrict__ x, half_t* __restrict__ xcp) {
  int idx = blockIdx.x * 256 + threadIdx.x;      // 524,288 threads, 4 w's each
  int w4 = idx & 31;                             // WW/4 = 32
  int h  = (idx >> 5) & (HH - 1);
  int d  = (idx >> 12) & (DD - 1);
  int b  = idx >> 18;

  const v4f* px = (const v4f*)x
      + ((size_t)b * CIN * DD + d) * (HH * WW / 4) + h * (WW / 4) + w4;
  const size_t cs = (size_t)DD * HH * WW / 4;    // channel stride in float4

  v4f t = __builtin_nontemporal_load(px);
  float sx = t[0], sy = t[1], sz = t[2], sw = t[3];
  float mx = t[0], my = t[1], mz = t[2], mw = t[3];
#pragma unroll 8
  for (int c = 1; c < CIN; ++c) {
    v4f v = __builtin_nontemporal_load(px + (size_t)c * cs);
    sx += v[0]; sy += v[1]; sz += v[2]; sw += v[3];
    mx = fmaxf(mx, v[0]); my = fmaxf(my, v[1]);
    mz = fmaxf(mz, v[2]); mw = fmaxf(mw, v[3]);
  }
  const float inv = 1.0f / (float)CIN;

  union { _Float16 h4[4]; unsigned long long u; } pa, pm;
  pa.h4[0] = (half_t)(sx * inv); pa.h4[1] = (half_t)(sy * inv);
  pa.h4[2] = (half_t)(sz * inv); pa.h4[3] = (half_t)(sw * inv);
  pm.h4[0] = (half_t)mx; pm.h4[1] = (half_t)my;
  pm.h4[2] = (half_t)mz; pm.h4[3] = (half_t)mw;

  size_t base = ((((size_t)b * 2 + 0) * DP + (d + DLEAD)) * HP + (h + HLEAD)) * WP
              + WLEAD + 4 * w4;                  // byte offset multiple of 8
  *(unsigned long long*)(xcp + base) = pa.u;                       // avg -> ch0
  *(unsigned long long*)(xcp + base + (size_t)DP * HP * WP) = pm.u; // max -> ch1
}

// ---------------------------------------------------------------------------
// Stage 2a: build Toeplitz A-fragments (one per (c,kd,kh)) in a contiguous-k (not AMD table order)
// 16-bit 16x32 A-matrix lane layout:
//   lane group g = lane>>4, row m = lane&15
//   V0..7 hold K = 16g + {0..15}
// A[m, k] = wt[k - m - 1] when 0 <= k-m-1 < 7, else 0.
// ---------------------------------------------------------------------------
__global__ void sa_wprep_kernel(const float* __restrict__ Wt,
                                unsigned int* __restrict__ afrag) {
  int f = blockIdx.x;                // 0..97 == (c*7+kd)*7 + kh
  int lane = threadIdx.x;            // 0..31
  float wt[7];
#pragma unroll
  for (int kw = 0; kw < 7; ++kw) wt[kw] = Wt[f * 7 + kw];

  int m = lane & 15, g = lane >> 4;
  unsigned int o[8];
#pragma unroll
  for (int v = 0; v < 8; ++v) {
    union { _Float16 h2[2]; unsigned int u; } pk;
#pragma unroll
    for (int hi = 0; hi < 2; ++hi) {
      int k = 16 * g + 2 * v + hi;
      int delta = k - m - 1;
      float val = (delta >= 0 && delta < 7) ? wt[delta] : 0.0f;
      pk.h2[hi] = (half_t)val;
    }
    o[v] = pk.u;
  }
  uint4* dst = (uint4*)(afrag + ((size_t)f * 32 + lane) * 8);
  dst[0] = make_uint4(o[0], o[1], o[2], o[3]);
  dst[1] = make_uint4(o[4], o[5], o[6], o[7]);
}

// ---------------------------------------------------------------------------
// Stage 2b: conv as implicit GEMM. One wave per 16x16 (h x w) output tile.
// Per (c,kd,kh): D[w,h] += A_toeplitz(16x32 f16) x B_data(32x16 f16).
// B lane layout: lanes 0-15 hold K=0..15 (n = lane), lanes 16-31 hold K=16..31
// (n = lane-16) -> each lane loads 16 contiguous halves of row h0+(lane&15)+kh.
// ---------------------------------------------------------------------------
__global__ void __launch_bounds__(256)
sa_conv_kernel(const half_t* __restrict__ xcp,
               const unsigned int* __restrict__ afragG,
               float* __restrict__ out) {
  __shared__ v16h frag_lds[49 * 32];            // 50,176 B: one channel's frags

  const int tid  = threadIdx.x;
  const int lane = tid & 31;
  const int wave = tid >> 5;
  const int tile = blockIdx.x * 8 + wave;       // 8192 tiles total
  const int w0 = (tile & 7) << 4;
  const int h0 = ((tile >> 3) & 7) << 4;
  const int d  = (tile >> 6) & (DD - 1);
  const int b  = tile >> 12;

  const int row = h0 + (lane & 15);             // B column n -> h row
  const int wb  = w0 + ((lane >> 4) << 4);      // K group start (32B aligned)

  v8f acc = {0.f, 0.f, 0.f, 0.f, 0.f, 0.f, 0.f, 0.f};

  for (int c = 0; c < 2; ++c) {
    __syncthreads();                            // protect LDS reuse across c
    {
      unsigned int* du = (unsigned int*)frag_lds;
      const unsigned int* su = afragG + (size_t)c * 49 * 256;
      for (int i = tid; i < 49 * 256; i += 256) du[i] = su[i];
    }
    __syncthreads();

    const half_t* cb = xcp + (((size_t)b * 2 + c) * DP + d) * HP * WP
                     + (size_t)row * WP + wb;
    for (int kd = 0; kd < 7; ++kd) {
      const half_t* bp = cb + (size_t)kd * (HP * WP);
#pragma unroll
      for (int kh = 0; kh < 7; ++kh) {
        v16h bfrag = *(const v16h*)(bp + kh * WP);
        v16h af = frag_lds[(kd * 7 + kh) * 32 + lane];
        acc = __builtin_amdgcn_wmma_f32_16x16x32_f16(
            /*neg_a=*/false, af, /*neg_b=*/false, bfrag,
            /*c_mod=*/(short)0, acc, /*reuse_a=*/false, /*reuse_b=*/false);
      }
    }
  }

  // D layout: n = lane&15 (h), M = VGPR r (+8 for lanes 16-31) = w offset.
  const int g = lane >> 4;
  const int h = h0 + (lane & 15);
  float vals[8];
#pragma unroll
  for (int r = 0; r < 8; ++r) vals[r] = 1.0f / (1.0f + __expf(-acc[r]));
  float* op = out + (((size_t)b * DD + d) * HH + h) * WW + w0 + 8 * g;
  *(float4*)op       = make_float4(vals[0], vals[1], vals[2], vals[3]);
  *((float4*)op + 1) = make_float4(vals[4], vals[5], vals[6], vals[7]);
}

// ---------------------------------------------------------------------------
extern "C" void kernel_launch(void* const* d_in, const int* in_sizes, int n_in,
                              void* d_out, int out_size, void* d_ws, size_t ws_size,
                              hipStream_t stream) {
  const float* x  = (const float*)d_in[0];   // [2,32,64,128,128] f32
  const float* Wt = (const float*)d_in[1];   // [1,2,7,7,7] f32 (686 flat)
  float* out = (float*)d_out;                // [2,1,64,128,128] f32

  if (ws_size < XCP_BYTES + FRAG_BYTES) return;
  half_t* xcp = (half_t*)d_ws;
  unsigned int* afrag = (unsigned int*)((char*)d_ws + XCP_BYTES);

  // Zero padded buffer (halos must be 0 every call; interior overwritten).
  hipMemsetAsync(d_ws, 0, XCP_BYTES, stream);

  sa_reduce_kernel<<<(BATCH * DD * HH * WW / 4) / 256, 256, 0, stream>>>(x, xcp);
  sa_wprep_kernel<<<NFRAG, 32, 0, stream>>>(Wt, afrag);
  sa_conv_kernel<<<(BATCH * DD * (HH / 16) * (WW / 16)) / 8, 256, 0, stream>>>(
      xcp, afrag, out);
}
